// CrossAttention_11132555231714
// MI455X (gfx1250) — matrix operand order repaired and checked by reading, not compiled, not run on hardware
//
#include <hip/hip_runtime.h>
#include <hip/hip_bf16.h>

// ---------------------------------------------------------------------------
// CrossAttention forward for MI455X (gfx1250), bf16 WMMA (16x16x32) pipeline.
// B=32, H=8, S=512, DH=64, D=512, Lq=384, Lk=448.
// d_out = [ out (B*S*D f32) | attn (B*H*S*S f32) ]
// ---------------------------------------------------------------------------

typedef __attribute__((ext_vector_type(16))) __bf16 v16bf;
typedef __attribute__((ext_vector_type(8)))  __bf16 v8bf;
typedef __attribute__((ext_vector_type(8)))  float  v8f;

#define BB   32
#define SS   512
#define DD   512
#define HH   8
#define DH   64
#define NEG_INF (-__builtin_huge_valf())

__device__ __forceinline__ v8f wmma_bf16(v16bf a, v16bf b, v8f c) {
  // D = A(16x32 bf16) x B(32x16 bf16) + C(16x16 f32)
  return __builtin_amdgcn_wmma_f32_16x16x32_bf16(
      false, a, false, b, (short)0, c, false, false);
}

__device__ __forceinline__ v16bf concat8(v8bf lo, v8bf hi) {
  return __builtin_shufflevector(lo, hi, 0,1,2,3,4,5,6,7,8,9,10,11,12,13,14,15);
}

__device__ __forceinline__ v16bf cvt16(float4 a, float4 b, float4 c, float4 d) {
  v16bf r;
  r[0]=(__bf16)a.x;  r[1]=(__bf16)a.y;  r[2]=(__bf16)a.z;  r[3]=(__bf16)a.w;
  r[4]=(__bf16)b.x;  r[5]=(__bf16)b.y;  r[6]=(__bf16)b.z;  r[7]=(__bf16)b.w;
  r[8]=(__bf16)c.x;  r[9]=(__bf16)c.y;  r[10]=(__bf16)c.z; r[11]=(__bf16)c.w;
  r[12]=(__bf16)d.x; r[13]=(__bf16)d.y; r[14]=(__bf16)d.z; r[15]=(__bf16)d.w;
  return r;
}

// B-fragment from a f32 weight row: element i -> K = k0 + (i<8 ? 0 : 16) + 8*hi + (i&7)
__device__ __forceinline__ v16bf load_wfrag(const float* __restrict__ wrow,
                                            int k0, int hi) {
  float4 w0 = *(const float4*)(wrow + k0 + 8*hi);
  float4 w1 = *(const float4*)(wrow + k0 + 8*hi + 4);
  float4 w2 = *(const float4*)(wrow + k0 + 16 + 8*hi);
  float4 w3 = *(const float4*)(wrow + k0 + 16 + 8*hi + 4);
  return cvt16(w0, w1, w2, w3);
}

// ---------------------------------------------------------------------------
// Kernel 1: P = pad(X) @ W^T + b, f32 in -> bf16 out.
// Each wave computes a 16x64 output strip: A loaded once per k-step,
// reused across 4 WMMAs (4 column tiles).
// vmode==0 : out[b][head][s][dh]   (for Q, K)
// vmode==1 : out[b][head][dh][s]   (transposed, for V)
// 1024 row-tiles x 8 col-groups = 8192 waves; 4 waves/block -> 2048 blocks.
// ---------------------------------------------------------------------------
__global__ __launch_bounds__(128)
void proj_f32_kernel(const float* __restrict__ X, int L,
                     const float* __restrict__ W, const float* __restrict__ bias,
                     unsigned short* __restrict__ outu, int vmode) {
  const int wave = threadIdx.x >> 5;
  const int lane = threadIdx.x & 31;
  const int m  = lane & 15;
  const int hi = lane >> 4;

  const int tile = blockIdx.x * 4 + wave;          // 0 .. 8191
  const int rt = tile & 1023;                      // row tile (B*S/16)
  const int cg = tile >> 10;                       // col group (D/64)
  const int s0g = rt * 16;
  const int b  = s0g >> 9;
  const int s0 = s0g & (SS - 1);
  const int d0 = cg * 64;

  const int  srow  = s0 + m;
  const bool valid = srow < L;                     // uniform (L multiple of 16)
  const float* xrow = X + (long)(b * L + srow) * DD;
  const float* wrow0 = W + (long)(d0 + m) * DD;    // B col n=m of tile c: row d0+16c+m
  const float* wrow1 = wrow0 + 16 * DD;
  const float* wrow2 = wrow0 + 32 * DD;
  const float* wrow3 = wrow0 + 48 * DD;

  v8f acc0 = {0.f,0.f,0.f,0.f,0.f,0.f,0.f,0.f};
  v8f acc1 = acc0, acc2 = acc0, acc3 = acc0;

  for (int k0 = 0; k0 < DD; k0 += 32) {
    v16bf a;
    if (valid) {
      // A element i -> K = k0 + (i<8 ? 0 : 16) + 8*hi + (i&7)
      float4 f0 = *(const float4*)(xrow + k0 + 8*hi);
      float4 f1 = *(const float4*)(xrow + k0 + 8*hi + 4);
      float4 f2 = *(const float4*)(xrow + k0 + 16 + 8*hi);
      float4 f3 = *(const float4*)(xrow + k0 + 16 + 8*hi + 4);
      a = cvt16(f0, f1, f2, f3);
    } else {
      float4 z = make_float4(0.f, 0.f, 0.f, 0.f);
      a = cvt16(z, z, z, z);
    }
    acc0 = wmma_bf16(a, load_wfrag(wrow0, k0, hi), acc0);
    acc1 = wmma_bf16(a, load_wfrag(wrow1, k0, hi), acc1);
    acc2 = wmma_bf16(a, load_wfrag(wrow2, k0, hi), acc2);
    acc3 = wmma_bf16(a, load_wfrag(wrow3, k0, hi), acc3);
  }

  // C/D layout: element r -> row M = r + 8*hi, col N = m
  __bf16* out = (__bf16*)outu;
  v8f accs[4] = {acc0, acc1, acc2, acc3};
  #pragma unroll
  for (int c = 0; c < 4; c++) {
    const int dcol = d0 + 16*c + m;
    const float bval = bias[dcol];
    const int head = dcol >> 6;
    const int dh   = dcol & (DH - 1);
    #pragma unroll
    for (int r = 0; r < 8; r++) {
      const int t = s0 + r + 8*hi;
      const float v = accs[c][r] + bval;
      long idx;
      if (vmode) idx = ((long)(b*HH + head) * DH + dh) * SS + t;    // [B,H,DH,S]
      else       idx = ((long)(b*HH + head) * SS + t) * DH + dh;    // [B,H,S,DH]
      out[idx] = (__bf16)v;
    }
  }
}

// ---------------------------------------------------------------------------
// Kernel 2: per (b, h, 16 query rows):
//   scores = (Q K^T)/8 * layer_scale[k], mask -> -inf, softmax (f32, LDS),
//   write attn f32, then ctx = attn @ V via bf16 WMMA (A from LDS).
// 256 threads = 8 waves. Waves 0..7 each do 4 score column-tiles (Q fragment
// hoisted, loaded once); waves 0..3 each do one 16x16 ctx tile.
// ---------------------------------------------------------------------------
__global__ __launch_bounds__(256)
void attn_kernel(const unsigned short* __restrict__ Qu,
                 const unsigned short* __restrict__ Ku,
                 const unsigned short* __restrict__ Vtu,
                 const int* __restrict__ mask,
                 const float* __restrict__ ls,
                 float* __restrict__ attn,
                 unsigned short* __restrict__ ctxu) {
  __shared__ __align__(16) float  sc[16 * SS];   // 32 KB scores / probs (f32)
  __shared__ __align__(16) __bf16 sp[16 * SS];   // 16 KB probs (bf16)

  const __bf16* Q  = (const __bf16*)Qu;
  const __bf16* K  = (const __bf16*)Ku;
  const __bf16* Vt = (const __bf16*)Vtu;

  const int tid  = threadIdx.x;
  const int wave = tid >> 5;
  const int lane = tid & 31;
  const int m  = lane & 15;
  const int hi = lane >> 4;

  const int qb = blockIdx.x & 31;
  const int h  = (blockIdx.x >> 5) & 7;
  const int b  = blockIdx.x >> 8;
  const int q0 = qb * 16;
  const long bh = (long)(b * HH + h);

  // ---- part 1: scores ----
  const __bf16* Qbase = Q + (bh * SS + q0 + m) * DH;       // A row = q0+m
  const v16bf aq0 = concat8(*(const v8bf*)(Qbase + 8*hi),
                            *(const v8bf*)(Qbase + 16 + 8*hi));
  const v16bf aq1 = concat8(*(const v8bf*)(Qbase + 32 + 8*hi),
                            *(const v8bf*)(Qbase + 48 + 8*hi));
  #pragma unroll
  for (int t = 0; t < 4; t++) {
    const int kt0 = (wave * 4 + t) * 16;
    const __bf16* Kbase = K + (bh * SS + kt0 + m) * DH;    // B col = key kt0+m
    v8f acc = {0.f,0.f,0.f,0.f,0.f,0.f,0.f,0.f};
    acc = wmma_bf16(aq0, concat8(*(const v8bf*)(Kbase + 8*hi),
                                 *(const v8bf*)(Kbase + 16 + 8*hi)), acc);
    acc = wmma_bf16(aq1, concat8(*(const v8bf*)(Kbase + 32 + 8*hi),
                                 *(const v8bf*)(Kbase + 48 + 8*hi)), acc);
    const int kt = kt0 + m;
    const float scale = 0.125f * ls[kt];                   // 1/sqrt(64)*layer_scale
    const bool msk = (mask[b * SS + kt] == 0);
    #pragma unroll
    for (int r = 0; r < 8; r++) {
      float v = acc[r] * scale;
      if (msk) v = NEG_INF;
      sc[(r + 8*hi) * SS + kt] = v;
    }
  }
  __syncthreads();

  // ---- part 2: softmax over each of the 16 rows (16 threads per row) ----
  const int row = tid >> 4;
  const int l16 = tid & 15;
  float* srow = sc + row * SS;
  float mx = NEG_INF;
  for (int c = l16; c < SS; c += 16) mx = fmaxf(mx, srow[c]);
  #pragma unroll
  for (int off = 8; off > 0; off >>= 1) mx = fmaxf(mx, __shfl_xor(mx, off, 16));
  float sum = 0.f;
  for (int c = l16; c < SS; c += 16) {
    float e = __expf(srow[c] - mx);
    srow[c] = e;
    sum += e;
  }
  #pragma unroll
  for (int off = 8; off > 0; off >>= 1) sum += __shfl_xor(sum, off, 16);
  const float inv = 1.f / sum;
  float* arow = attn + (bh * SS + q0 + row) * SS;
  __bf16* prow = sp + row * SS;
  for (int c = l16; c < SS; c += 16) {
    float p = srow[c] * inv;
    arow[c] = p;                 // f32 attention output
    prow[c] = (__bf16)p;         // bf16 for the AV matmul
  }
  __syncthreads();

  // ---- part 3: ctx = attn @ V  (waves 0..3, one 16x16 tile each) ----
  if (wave < 4) {
    const int dh0 = wave * 16;
    const __bf16* Vbase = Vt + (bh * DH + dh0 + m) * SS;   // B col = dh0+m
    const __bf16* Abase = sp + m * SS;                     // A row = m
    v8f acc = {0.f,0.f,0.f,0.f,0.f,0.f,0.f,0.f};
    for (int kk = 0; kk < SS; kk += 32) {
      v16bf a  = concat8(*(const v8bf*)(Abase + kk + 8*hi),
                         *(const v8bf*)(Abase + kk + 16 + 8*hi));
      v16bf bm = concat8(*(const v8bf*)(Vbase + kk + 8*hi),
                         *(const v8bf*)(Vbase + kk + 16 + 8*hi));
      acc = wmma_bf16(a, bm, acc);
    }
    __bf16* ctx = (__bf16*)ctxu;
    #pragma unroll
    for (int r = 0; r < 8; r++) {
      const int q = q0 + r + 8*hi;
      ctx[((long)(b * SS + q)) * DD + h * DH + dh0 + m] = (__bf16)acc[r];
    }
  }
}

// ---------------------------------------------------------------------------
// Kernel 3: out = ctx(bf16) @ Wo^T + bo, f32 out. 16x64 strip per wave.
// ---------------------------------------------------------------------------
__global__ __launch_bounds__(128)
void proj_bf16_kernel(const unsigned short* __restrict__ Xu,
                      const float* __restrict__ W, const float* __restrict__ bias,
                      float* __restrict__ out) {
  const __bf16* X = (const __bf16*)Xu;
  const int wave = threadIdx.x >> 5;
  const int lane = threadIdx.x & 31;
  const int m  = lane & 15;
  const int hi = lane >> 4;

  const int tile = blockIdx.x * 4 + wave;          // 0 .. 8191
  const int rt = tile & 1023;
  const int cg = tile >> 10;
  const int s0g = rt * 16;
  const int d0  = cg * 64;

  const __bf16* xrow = X + (long)(s0g + m) * DD;
  const float* wrow0 = W + (long)(d0 + m) * DD;
  const float* wrow1 = wrow0 + 16 * DD;
  const float* wrow2 = wrow0 + 32 * DD;
  const float* wrow3 = wrow0 + 48 * DD;

  v8f acc0 = {0.f,0.f,0.f,0.f,0.f,0.f,0.f,0.f};
  v8f acc1 = acc0, acc2 = acc0, acc3 = acc0;

  for (int k0 = 0; k0 < DD; k0 += 32) {
    v16bf a = concat8(*(const v8bf*)(xrow + k0 + 8*hi),
                      *(const v8bf*)(xrow + k0 + 16 + 8*hi));
    acc0 = wmma_bf16(a, load_wfrag(wrow0, k0, hi), acc0);
    acc1 = wmma_bf16(a, load_wfrag(wrow1, k0, hi), acc1);
    acc2 = wmma_bf16(a, load_wfrag(wrow2, k0, hi), acc2);
    acc3 = wmma_bf16(a, load_wfrag(wrow3, k0, hi), acc3);
  }

  v8f accs[4] = {acc0, acc1, acc2, acc3};
  #pragma unroll
  for (int c = 0; c < 4; c++) {
    const int dcol = d0 + 16*c + m;
    const float bval = bias[dcol];
    #pragma unroll
    for (int r = 0; r < 8; r++) {
      out[(long)(s0g + r + 8*hi) * DD + dcol] = accs[c][r] + bval;
    }
  }
}

// ---------------------------------------------------------------------------
extern "C" void kernel_launch(void* const* d_in, const int* in_sizes, int n_in,
                              void* d_out, int out_size, void* d_ws, size_t ws_size,
                              hipStream_t stream) {
  const float* query = (const float*)d_in[0];   // [32,384,512]
  const float* key   = (const float*)d_in[1];   // [32,448,512]
  const float* value = (const float*)d_in[2];   // [32,448,512]
  const int*   mask  = (const int*)  d_in[3];   // [32,512]
  const float* Wq = (const float*)d_in[4];
  const float* bq = (const float*)d_in[5];
  const float* Wk = (const float*)d_in[6];
  const float* bk = (const float*)d_in[7];
  const float* Wv = (const float*)d_in[8];
  const float* bv = (const float*)d_in[9];
  const float* Wo = (const float*)d_in[10];
  const float* bo = (const float*)d_in[11];
  const float* ls = (const float*)d_in[12];

  float* out  = (float*)d_out;                          // B*S*D
  float* attn = out + (long)BB * SS * DD;               // B*H*S*S

  const long NE = (long)BB * HH * SS * DH;              // 8,388,608 bf16 elems
  unsigned short* ws = (unsigned short*)d_ws;
  unsigned short* Qw  = ws;                             // [B,H,S,DH] bf16
  unsigned short* Kw  = ws + NE;                        // [B,H,S,DH] bf16
  unsigned short* Vtw = ws + 2 * NE;                    // [B,H,DH,S] bf16
  unsigned short* Cw  = ws + 3 * NE;                    // [B,S,D]    bf16

  const dim3 gproj(2048), bproj(128);                   // 8192 strips, 4 waves/blk
  proj_f32_kernel<<<gproj, bproj, 0, stream>>>(query, 384, Wq, bq, Qw, 0);
  proj_f32_kernel<<<gproj, bproj, 0, stream>>>(key,   448, Wk, bk, Kw, 0);
  proj_f32_kernel<<<gproj, bproj, 0, stream>>>(value, 448, Wv, bv, Vtw, 1);

  attn_kernel<<<dim3(BB * HH * (SS / 16)), dim3(256), 0, stream>>>(
      Qw, Kw, Vtw, mask, ls, attn, Cw);

  proj_bf16_kernel<<<gproj, bproj, 0, stream>>>(Cw, Wo, bo, out);
}
